// ViT_34857954574665
// MI455X (gfx1250) — matrix operand order repaired and checked by reading, not compiled, not run on hardware
//
#include <hip/hip_runtime.h>

// ---------------- types ----------------
typedef __attribute__((ext_vector_type(16))) __bf16 v16bf;
typedef __attribute__((ext_vector_type(8)))  float  v8f;

#define L_  12
#define E_  768
#define H_  12
#define D_  64
#define FC_ 3072
#define B_  32
#define S_  197
#define SP_ 208            // padded S (13*16)
#define PRW_ 112           // packed prob row stride in dwords (224 bf16 cols, zero padded)
#define BS_ (B_ * S_)      // 6304
#define NPATCH_ (B_ * 196)

// fp32 -> bf16 bits (round-to-nearest-even-ish)
static __device__ __forceinline__ unsigned bfbits(float f) {
    union { float f; unsigned u; } x; x.f = f;
    return (x.u + 0x7FFFu + ((x.u >> 16) & 1u)) >> 16;
}
static __device__ __forceinline__ unsigned pk2(float lo, float hi) {
    return (bfbits(lo) & 0xFFFFu) | (bfbits(hi) << 16);
}
static __device__ __forceinline__ int imin(int a, int b) { return a < b ? a : b; }

union Frag { v16bf v; unsigned u[8]; };

static __device__ __forceinline__ v8f wmma_bf16(v16bf a, v16bf b, v8f c) {
    return __builtin_amdgcn_wmma_f32_16x16x32_bf16(
        false, a, false, b, (short)0, c, false, false);
}

// ---- CDNA5 async global->LDS copy (ASYNCcnt-tracked, bypasses VGPRs) ----
// GV mode: per-lane LDS byte address in VDST VGPR, 64-bit global address in VADDR.
static __device__ __forceinline__ void async_ld_b128(unsigned lds_off, const void* gaddr) {
    asm volatile("global_load_async_to_lds_b128 %0, %1, off"
                 :: "v"(lds_off), "v"(gaddr) : "memory");
}
static __device__ __forceinline__ void wait_async0() {
    asm volatile("s_wait_asynccnt 0" ::: "memory");
}

// ---------------- generic WMMA GEMM (bf16 in, fp32 acc) ----------------
// C = act(A @ B^T-layout + bias). A: bf16 [M][K]; Bt: bf16 [N][K] (pre-transposed).
// Block 256 thr (8 waves), tile 64(M) x 128(N), wave 32x32, K-step 32.
// Tiles staged with async b128 global->LDS copies; no conversions or masks in loop.
template <int ACT, int OUTBF>   // ACT: 0=none 1=GELU; OUTBF: 0=f32 out, 1=bf16 out
__global__ __launch_bounds__(256) void gemm_wmma(
    const unsigned short* __restrict__ A, const unsigned short* __restrict__ Bt,
    const float* __restrict__ bias, void* __restrict__ Cout,
    int M, int N, int K)
{
    __shared__ __align__(16) unsigned As32[64 * 16];    // [row][k/2]
    __shared__ __align__(16) unsigned Bst32[128 * 16];  // [col][k/2]

    const int tid  = threadIdx.x;
    const int lane = tid & 31;
    const int wid  = tid >> 5;
    const int wm0  = (wid >> 2) * 32;
    const int wn0  = (wid & 3) * 32;
    const int bm0  = blockIdx.y * 64;
    const int bn0  = blockIdx.x * 128;

    const unsigned* Ad = reinterpret_cast<const unsigned*>(A);
    const unsigned* Bd = reinterpret_cast<const unsigned*>(Bt);
    const int K2 = K >> 1;

    v8f acc00 = {}, acc01 = {}, acc10 = {}, acc11 = {};

    const int arow0 = wm0 + (lane & 15);
    const int a2b   = (lane < 16) ? 0 : 8;   // akb/2
    const int b2b   = (lane < 16) ? 0 : 8;   // bkb/2
    const int bcol0 = wn0 + (lane & 15);

    // staging coordinates (per thread: one/two uint4 chunks)
    const int arl = tid >> 2, aq4 = (tid & 3) * 4;
    const int agr = imin(bm0 + arl, M - 1);
    const int c1t = tid + 256;
    const int bn_l0 = tid >> 2, bq4_0 = (tid & 3) * 4;
    const int bn_l1 = c1t >> 2, bq4_1 = (c1t & 3) * 4;

    // loop-invariant LDS destination byte addresses
    const unsigned ldsA  = (unsigned)(uintptr_t)(&As32[arl * 16 + aq4]);
    const unsigned ldsB0 = (unsigned)(uintptr_t)(&Bst32[bn_l0 * 16 + bq4_0]);
    const unsigned ldsB1 = (unsigned)(uintptr_t)(&Bst32[bn_l1 * 16 + bq4_1]);

    const unsigned* gA  = Ad + (size_t)agr * K2 + aq4;
    const unsigned* gB0 = Bd + (size_t)(bn0 + bn_l0) * K2 + bq4_0;
    const unsigned* gB1 = Bd + (size_t)(bn0 + bn_l1) * K2 + bq4_1;

    for (int k0 = 0; k0 < K; k0 += 32) {
        const int kd = k0 >> 1;
        // ---- async stage A (64x16 dwords) and B (128x16 dwords) tiles ----
        async_ld_b128(ldsA,  gA  + kd);
        async_ld_b128(ldsB0, gB0 + kd);
        async_ld_b128(ldsB1, gB1 + kd);
        wait_async0();
        __syncthreads();

        // ---- fragments: uint4 LDS reads ----
        Frag a0, a1, b0, b1;
        const int ab0 = arow0 * 16 + a2b;
        const int ab1 = (arow0 + 16) * 16 + a2b;
        *reinterpret_cast<uint4*>(&a0.u[0]) = *reinterpret_cast<const uint4*>(&As32[ab0]);
        *reinterpret_cast<uint4*>(&a0.u[4]) = *reinterpret_cast<const uint4*>(&As32[ab0 + 4]);
        *reinterpret_cast<uint4*>(&a1.u[0]) = *reinterpret_cast<const uint4*>(&As32[ab1]);
        *reinterpret_cast<uint4*>(&a1.u[4]) = *reinterpret_cast<const uint4*>(&As32[ab1 + 4]);
        const int bb0 = bcol0 * 16 + b2b;
        const int bb1 = (bcol0 + 16) * 16 + b2b;
        *reinterpret_cast<uint4*>(&b0.u[0]) = *reinterpret_cast<const uint4*>(&Bst32[bb0]);
        *reinterpret_cast<uint4*>(&b0.u[4]) = *reinterpret_cast<const uint4*>(&Bst32[bb0 + 4]);
        *reinterpret_cast<uint4*>(&b1.u[0]) = *reinterpret_cast<const uint4*>(&Bst32[bb1]);
        *reinterpret_cast<uint4*>(&b1.u[4]) = *reinterpret_cast<const uint4*>(&Bst32[bb1 + 4]);

        acc00 = wmma_bf16(a0.v, b0.v, acc00);
        acc01 = wmma_bf16(a0.v, b1.v, acc01);
        acc10 = wmma_bf16(a1.v, b0.v, acc10);
        acc11 = wmma_bf16(a1.v, b1.v, acc11);
        __syncthreads();
    }

    const int srm = bm0 + wm0 + ((lane < 16) ? 0 : 8);
    const int scn = bn0 + wn0 + (lane & 15);
    #pragma unroll
    for (int t = 0; t < 4; ++t) {
        const v8f& acc = (t == 0) ? acc00 : (t == 1) ? acc01 : (t == 2) ? acc10 : acc11;
        const int rowb = srm + ((t >> 1) ? 16 : 0);
        const int col  = scn + ((t & 1) ? 16 : 0);
        float bv = bias ? bias[col] : 0.0f;
        #pragma unroll
        for (int r = 0; r < 8; ++r) {
            int row = rowb + r;
            if (row < M) {
                float v = acc[r] + bv;
                if (ACT == 1) v = 0.5f * v * (1.0f + erff(v * 0.70710678118654752f));
                if (OUTBF)
                    ((unsigned short*)Cout)[(size_t)row * N + col] = (unsigned short)bfbits(v);
                else
                    ((float*)Cout)[(size_t)row * N + col] = v;
            }
        }
    }
}

// ---------------- attention: logits = 0.125 * Q K^T (bf16 in, f32 out) ----------------
// grid (13 m-tiles, B*H), block 32. logits[bh][208][208], fully unguarded stores.
__global__ __launch_bounds__(32) void attn_qk(
    const unsigned short* __restrict__ q, const unsigned short* __restrict__ k,
    float* __restrict__ logits)
{
    const int bh = blockIdx.y, b = bh / H_, hh = bh % H_;
    const int lane = threadIdx.x;
    const int m0 = blockIdx.x * 16;
    const unsigned* Qd = reinterpret_cast<const unsigned*>(q) +
                         ((size_t)(b * S_) * E_ + hh * D_) / 2;
    const unsigned* Kd = reinterpret_cast<const unsigned*>(k) +
                         ((size_t)(b * S_) * E_ + hh * D_) / 2;

    const int a2b = (lane < 16) ? 0 : 8;   // akb/2
    const int b2b = (lane < 16) ? 0 : 8;   // bkb/2

    // A fragments: row-clamped, unconditional uint4 loads (pad rows are stored but never read)
    const unsigned* Qr = Qd + (size_t)imin(m0 + (lane & 15), S_ - 1) * (E_ / 2);
    Frag a[2];
    #pragma unroll
    for (int ks = 0; ks < 2; ++ks) {
        *reinterpret_cast<uint4*>(&a[ks].u[0]) =
            *reinterpret_cast<const uint4*>(Qr + ks * 16 + a2b);
        *reinterpret_cast<uint4*>(&a[ks].u[4]) =
            *reinterpret_cast<const uint4*>(Qr + ks * 16 + a2b + 4);
    }

    for (int nt = 0; nt < 13; ++nt) {
        const int n0 = nt * 16;
        const unsigned* Kr = Kd + (size_t)imin(n0 + (lane & 15), S_ - 1) * (E_ / 2);
        v8f acc = {};
        #pragma unroll
        for (int ks = 0; ks < 2; ++ks) {
            Frag bf;
            *reinterpret_cast<uint4*>(&bf.u[0]) =
                *reinterpret_cast<const uint4*>(Kr + ks * 16 + b2b);
            *reinterpret_cast<uint4*>(&bf.u[4]) =
                *reinterpret_cast<const uint4*>(Kr + ks * 16 + b2b + 4);
            acc = wmma_bf16(a[ks].v, bf.v, acc);
        }
        const int rowb = m0 + ((lane < 16) ? 0 : 8);
        const int col  = n0 + (lane & 15);
        float* Lb = logits + ((size_t)bh * SP_ + rowb) * SP_ + col;
        #pragma unroll
        for (int r = 0; r < 8; ++r)
            Lb[(size_t)r * SP_] = acc[r] * 0.125f;
    }
}

// ---------------- softmax: f32 logits row -> packed bf16 prob row (zero padded) ----------------
__global__ __launch_bounds__(128) void softmax_k(
    const float* __restrict__ logits, unsigned* __restrict__ probs)
{
    const int bh = blockIdx.x / S_;
    const int i  = blockIdx.x % S_;
    const float* Lr = logits + ((size_t)bh * SP_ + i) * SP_;
    unsigned* Pr = probs + ((size_t)bh * S_ + i) * PRW_;
    const int t = threadIdx.x;
    const int c0 = 2 * t, c1 = 2 * t + 1;

    float v0 = (c0 < S_) ? Lr[c0] : -1e30f;
    float v1 = (c1 < S_) ? Lr[c1] : -1e30f;

    __shared__ float red[128];
    red[t] = fmaxf(v0, v1);
    __syncthreads();
    for (int off = 64; off > 0; off >>= 1) {
        if (t < off) red[t] = fmaxf(red[t], red[t + off]);
        __syncthreads();
    }
    const float mx = red[0];
    __syncthreads();

    float e0 = __expf(v0 - mx);
    float e1 = __expf(v1 - mx);
    red[t] = e0 + e1;
    __syncthreads();
    for (int off = 64; off > 0; off >>= 1) {
        if (t < off) red[t] += red[t + off];
        __syncthreads();
    }
    const float inv = 1.0f / red[0];
    if (t < PRW_) Pr[t] = pk2(e0 * inv, e1 * inv);   // pad cols pack to exact 0
}

// ---------------- attention: O = P V (packed bf16 in, bf16 out) ----------------
__global__ __launch_bounds__(32) void attn_av(
    const unsigned* __restrict__ probs, const unsigned short* __restrict__ v,
    unsigned short* __restrict__ o)
{
    __shared__ __align__(16) unsigned Vst[64 * 16];   // [col][kpair]

    const int bh = blockIdx.y, b = bh / H_, hh = bh % H_;
    const int lane = threadIdx.x;
    const int m0 = blockIdx.x * 16;
    const unsigned* Vd = reinterpret_cast<const unsigned*>(v) +
                         ((size_t)(b * S_) * E_ + hh * D_) / 2;
    const unsigned* Pr = probs +
        (size_t)((size_t)bh * S_ + imin(m0 + (lane & 15), S_ - 1)) * PRW_;

    const int a2b = (lane < 16) ? 0 : 8;   // akb/2
    const int b2b = (lane < 16) ? 0 : 8;   // bkb/2

    v8f acc[4] = {{}, {}, {}, {}};

    for (int kt = 0; kt < 7; ++kt) {
        const int k0 = kt * 32;

        // stage V (32 x 64) transposed: chunk = (kpair, 4-col group), 8 per lane
        #pragma unroll
        for (int i = 0; i < 8; ++i) {
            int ch = lane + i * 32;
            int kp = ch >> 4, cg = ch & 15;
            int col0 = cg * 4;
            int r0 = imin(k0 + 2 * kp, S_ - 1);
            int r1 = imin(k0 + 2 * kp + 1, S_ - 1);
            uint2 d0 = *reinterpret_cast<const uint2*>(Vd + (size_t)r0 * (E_ / 2) + (col0 >> 1));
            uint2 d1 = *reinterpret_cast<const uint2*>(Vd + (size_t)r1 * (E_ / 2) + (col0 >> 1));
            Vst[(col0 + 0) * 16 + kp] = (d0.x & 0xFFFFu) | (d1.x << 16);
            Vst[(col0 + 1) * 16 + kp] = (d0.x >> 16) | (d1.x & 0xFFFF0000u);
            Vst[(col0 + 2) * 16 + kp] = (d0.y & 0xFFFFu) | (d1.y << 16);
            Vst[(col0 + 3) * 16 + kp] = (d0.y >> 16) | (d1.y & 0xFFFF0000u);
        }
        __syncthreads();

        // P fragment: unconditional uint4 loads (pad cols are exact zeros)
        Frag a;
        *reinterpret_cast<uint4*>(&a.u[0]) =
            *reinterpret_cast<const uint4*>(Pr + kt * 16 + a2b);
        *reinterpret_cast<uint4*>(&a.u[4]) =
            *reinterpret_cast<const uint4*>(Pr + kt * 16 + a2b + 4);

        #pragma unroll
        for (int nf = 0; nf < 4; ++nf) {
            const int col = nf * 16 + (lane & 15);
            Frag bf;
            const int bb = col * 16 + b2b;
            *reinterpret_cast<uint4*>(&bf.u[0]) = *reinterpret_cast<const uint4*>(&Vst[bb]);
            *reinterpret_cast<uint4*>(&bf.u[4]) = *reinterpret_cast<const uint4*>(&Vst[bb + 4]);
            acc[nf] = wmma_bf16(a.v, bf.v, acc[nf]);
        }
        __syncthreads();
    }

    const int rowb = m0 + ((lane < 16) ? 0 : 8);
    #pragma unroll
    for (int nf = 0; nf < 4; ++nf) {
        const int col = nf * 16 + (lane & 15);
        #pragma unroll
        for (int r = 0; r < 8; ++r) {
            int row = rowb + r;
            if (row < S_)
                o[(size_t)(b * S_ + row) * E_ + hh * D_ + col] =
                    (unsigned short)bfbits(acc[nf][r]);
        }
    }
}

// ---------------- weight convert + transpose: f32 [K][N] -> bf16 [N][K] ----------------
__global__ void wconv_k(const float* __restrict__ w, unsigned short* __restrict__ wt,
                        int K, int N) {
    size_t idx = (size_t)blockIdx.x * 256 + threadIdx.x;
    if (idx >= (size_t)K * N) return;
    int n = (int)(idx / K), kk = (int)(idx % K);
    wt[idx] = (unsigned short)bfbits(w[(size_t)kk * N + n]);
}

// ---------------- patchify: img f32 -> Xp bf16 [B*196][768] ----------------
__global__ void patchify_k(const float* __restrict__ img, unsigned short* __restrict__ xp) {
    size_t idx = (size_t)blockIdx.x * blockDim.x + threadIdx.x;
    if (idx >= (size_t)NPATCH_ * E_) return;
    int col = (int)(idx % E_);
    int row = (int)(idx / E_);
    int b = row / 196, p = row % 196;
    int ph = p / 14, pw = p % 14;
    int ch = col % 3, pix = col / 3;
    int p1 = pix / 16, p2 = pix % 16;
    float v = img[(((size_t)b * 3 + ch) * 224 + (ph * 16 + p1)) * 224 + (pw * 16 + p2)];
    xp[idx] = (unsigned short)bfbits(v);
}

// ---------------- cls + 2D sincos pos: tokens f32 -> x (f32 + bf16) ----------------
__global__ void clspos_k(const float* __restrict__ tok, const float* __restrict__ cls,
                         float* __restrict__ xf, unsigned short* __restrict__ xh) {
    size_t idx = (size_t)blockIdx.x * blockDim.x + threadIdx.x;
    if (idx >= (size_t)BS_ * E_) return;
    int col = (int)(idx % E_);
    int row = (int)(idx / E_);
    int b = row / S_, s = row % S_;
    float v;
    if (s == 0) {
        v = cls[col];
    } else {
        int p = s - 1;
        int py = p / 14, px = p % 14;
        int quarter = col / 192, j = col % 192;
        float om = powf(10000.0f, -(float)j / 191.0f);
        float arg = ((quarter < 2) ? (float)px : (float)py) * om;
        float pe = ((quarter & 1) == 0) ? sinf(arg) : cosf(arg);
        v = tok[(size_t)(b * 196 + p) * E_ + col] + pe;
    }
    xf[idx] = v;
    xh[idx] = (unsigned short)bfbits(v);
}

// ---------------- LayerNorm (optional residual), dual f32/bf16 outputs ----------------
__global__ __launch_bounds__(256) void ln_k(
    float* __restrict__ outf, unsigned short* __restrict__ outh,
    const float* __restrict__ a, const float* __restrict__ res,
    const float* __restrict__ g, const float* __restrict__ bta)
{
    const int row = blockIdx.x;
    const int tid = threadIdx.x;
    const float* pa = a + (size_t)row * E_;
    const float* pr = res ? res + (size_t)row * E_ : nullptr;

    float v[3], s = 0.f, s2 = 0.f;
    #pragma unroll
    for (int i = 0; i < 3; ++i) {
        int c = tid + i * 256;
        float x = pa[c] + (pr ? pr[c] : 0.f);
        v[i] = x; s += x; s2 += x * x;
    }
    __shared__ float rs[256], rq[256];
    rs[tid] = s; rq[tid] = s2;
    __syncthreads();
    for (int off = 128; off > 0; off >>= 1) {
        if (tid < off) { rs[tid] += rs[tid + off]; rq[tid] += rq[tid + off]; }
        __syncthreads();
    }
    float mean = rs[0] * (1.0f / E_);
    float var  = rq[0] * (1.0f / E_) - mean * mean;
    float inv  = rsqrtf(var + 1e-5f);
    #pragma unroll
    for (int i = 0; i < 3; ++i) {
        int c = tid + i * 256;
        float y = (v[i] - mean) * inv * g[c] + bta[c];
        if (outf) outf[(size_t)row * E_ + c] = y;
        if (outh) outh[(size_t)row * E_ + c] = (unsigned short)bfbits(y);
    }
}

// ---------------- classifier ----------------
__global__ void classifier_k(const float* __restrict__ x, const float* __restrict__ w,
                             const float* __restrict__ bias, float* __restrict__ out) {
    int t = threadIdx.x;
    if (t >= B_ * 2) return;
    int b = t >> 1, c = t & 1;
    const float* row = x + (size_t)(b * S_) * E_;
    float s = bias[c];
    for (int e = 0; e < E_; ++e) s += row[e] * w[e * 2 + c];
    out[b * 2 + c] = s;
}

// ---------------- host orchestration ----------------
extern "C" void kernel_launch(void* const* d_in, const int* in_sizes, int n_in,
                              void* d_out, int out_size, void* d_ws, size_t ws_size,
                              hipStream_t stream) {
    (void)in_sizes; (void)n_in; (void)out_size; (void)ws_size;
    const float* img     = (const float*)d_in[0];
    const float* patch_w = (const float*)d_in[1];
    const float* patch_b = (const float*)d_in[2];
    const float* cls_tok = (const float*)d_in[3];
    const float* Wq      = (const float*)d_in[4];
    const float* Wk      = (const float*)d_in[5];
    const float* Wv      = (const float*)d_in[6];
    const float* Wo      = (const float*)d_in[7];
    const float* bo      = (const float*)d_in[8];
    const float* ln1_g   = (const float*)d_in[9];
    const float* ln1_b   = (const float*)d_in[10];
    const float* fln_g   = (const float*)d_in[11];
    const float* fln_b   = (const float*)d_in[12];
    const float* fc1_w   = (const float*)d_in[13];
    const float* fc1_b   = (const float*)d_in[14];
    const float* fc2_w   = (const float*)d_in[15];
    const float* fc2_b   = (const float*)d_in[16];
    const float* ln2_g   = (const float*)d_in[17];
    const float* ln2_b   = (const float*)d_in[18];
    const float* clf_w   = (const float*)d_in[19];
    const float* clf_b   = (const float*)d_in[20];
    float* out = (float*)d_out;

    // ---- workspace carve (256B aligned chunks) ----
    char* wp = (char*)d_ws;
    auto carveF = [&](size_t n) { float* r = (float*)wp; wp += ((n * 4 + 255) / 256) * 256; return r; };
    auto carveH = [&](size_t n) { unsigned short* r = (unsigned short*)wp; wp += ((n * 2 + 255) / 256) * 256; return r; };
    auto carveU = [&](size_t n) { unsigned* r = (unsigned*)wp; wp += ((n * 4 + 255) / 256) * 256; return r; };

    const size_t NX = (size_t)BS_ * E_;
    float* xb           = carveF(NX);
    float* xn           = carveF(NX);
    float* logits       = carveF((size_t)B_ * H_ * SP_ * SP_);
    unsigned short* xbh = carveH(NX);
    unsigned short* xnh = carveH(NX);
    unsigned short* qh  = carveH(NX);
    unsigned short* kh  = carveH(NX);
    unsigned short* vh  = carveH(NX);
    unsigned short* obh = carveH(NX);
    unsigned short* hbh = carveH((size_t)BS_ * FC_);
    unsigned* probs     = carveU((size_t)B_ * H_ * S_ * PRW_ + 16);
    unsigned short* wtp = carveH((size_t)E_ * E_);
    unsigned short* wtq = carveH((size_t)L_ * E_ * E_);
    unsigned short* wtk = carveH((size_t)L_ * E_ * E_);
    unsigned short* wtv = carveH((size_t)L_ * E_ * E_);
    unsigned short* wto = carveH((size_t)L_ * E_ * E_);
    unsigned short* wt1 = carveH((size_t)L_ * E_ * FC_);
    unsigned short* wt2 = carveH((size_t)L_ * E_ * FC_);

    const dim3 blk256(256), blk128(128), blk32(32);
    const dim3 gAttn(13, B_ * H_);
    const unsigned gEE  = (unsigned)(((size_t)E_ * E_ + 255) / 256);
    const unsigned gEF  = (unsigned)(((size_t)E_ * FC_ + 255) / 256);

    // ---- one-time weight convert + transpose ----
    wconv_k<<<dim3(gEE), blk256, 0, stream>>>(patch_w, wtp, E_, E_);
    for (int l = 0; l < L_; ++l) {
        wconv_k<<<dim3(gEE), blk256, 0, stream>>>(Wq + (size_t)l * E_ * E_, wtq + (size_t)l * E_ * E_, E_, E_);
        wconv_k<<<dim3(gEE), blk256, 0, stream>>>(Wk + (size_t)l * E_ * E_, wtk + (size_t)l * E_ * E_, E_, E_);
        wconv_k<<<dim3(gEE), blk256, 0, stream>>>(Wv + (size_t)l * E_ * E_, wtv + (size_t)l * E_ * E_, E_, E_);
        wconv_k<<<dim3(gEE), blk256, 0, stream>>>(Wo + (size_t)l * E_ * E_, wto + (size_t)l * E_ * E_, E_, E_);
        wconv_k<<<dim3(gEF), blk256, 0, stream>>>(fc1_w + (size_t)l * E_ * FC_, wt1 + (size_t)l * E_ * FC_, E_, FC_);
        wconv_k<<<dim3(gEF), blk256, 0, stream>>>(fc2_w + (size_t)l * E_ * FC_, wt2 + (size_t)l * E_ * FC_, FC_, E_);
    }

    // ---- patch embed ----
    {
        size_t tot = (size_t)NPATCH_ * E_;
        patchify_k<<<dim3((unsigned)((tot + 255) / 256)), blk256, 0, stream>>>(img, hbh);
        gemm_wmma<0, 0><<<dim3(E_ / 128, (NPATCH_ + 63) / 64), blk256, 0, stream>>>(
            hbh, wtp, patch_b, xn, NPATCH_, E_, E_);
        size_t tot2 = (size_t)BS_ * E_;
        clspos_k<<<dim3((unsigned)((tot2 + 255) / 256)), blk256, 0, stream>>>(xn, cls_tok, xb, xbh);
    }

    // ---- transformer blocks ----
    for (int l = 0; l < L_; ++l) {
        const unsigned short* wq = wtq + (size_t)l * E_ * E_;
        const unsigned short* wk = wtk + (size_t)l * E_ * E_;
        const unsigned short* wv = wtv + (size_t)l * E_ * E_;
        const unsigned short* wo = wto + (size_t)l * E_ * E_;
        const unsigned short* w1 = wt1 + (size_t)l * E_ * FC_;
        const unsigned short* w2 = wt2 + (size_t)l * E_ * FC_;
        const float* bo_ = bo + (size_t)l * E_;
        const float* g1 = ln1_g + (size_t)l * E_;
        const float* b1 = ln1_b + (size_t)l * E_;
        const float* gf = fln_g + (size_t)l * E_;
        const float* bf = fln_b + (size_t)l * E_;
        const float* bw1 = fc1_b + (size_t)l * FC_;
        const float* bw2 = fc2_b + (size_t)l * E_;
        const float* g2 = ln2_g + (size_t)l * E_;
        const float* b2 = ln2_b + (size_t)l * E_;

        const dim3 gE(E_ / 128, (BS_ + 63) / 64);
        gemm_wmma<0, 1><<<gE, blk256, 0, stream>>>(xbh, wq, nullptr, qh, BS_, E_, E_);
        gemm_wmma<0, 1><<<gE, blk256, 0, stream>>>(xbh, wk, nullptr, kh, BS_, E_, E_);
        gemm_wmma<0, 1><<<gE, blk256, 0, stream>>>(xbh, wv, nullptr, vh, BS_, E_, E_);

        attn_qk<<<gAttn, blk32, 0, stream>>>(qh, kh, logits);
        softmax_k<<<dim3(B_ * H_ * S_), blk128, 0, stream>>>(logits, probs);
        attn_av<<<gAttn, blk32, 0, stream>>>(probs, vh, obh);

        gemm_wmma<0, 0><<<gE, blk256, 0, stream>>>(obh, wo, bo_, xn, BS_, E_, E_);
        ln_k<<<dim3(BS_), blk256, 0, stream>>>(xb, xbh, xn, xb, g1, b1);     // x = LN(o + x)
        ln_k<<<dim3(BS_), blk256, 0, stream>>>(nullptr, xnh, xb, nullptr, gf, bf);

        gemm_wmma<1, 1><<<dim3(FC_ / 128, (BS_ + 63) / 64), blk256, 0, stream>>>(
            xnh, w1, bw1, hbh, BS_, FC_, E_);                                 // GELU fused
        gemm_wmma<0, 0><<<gE, blk256, 0, stream>>>(hbh, w2, bw2, xn, BS_, E_, FC_);
        ln_k<<<dim3(BS_), blk256, 0, stream>>>(xb, xbh, xn, xb, g2, b2);     // x = LN(h + x)
    }

    classifier_k<<<dim3(1), dim3(64), 0, stream>>>(xb, clf_w, clf_b, out);
}
